// PoseEncoderGCN_40046275068261
// MI455X (gfx1250) — compile-verified, matrix operand order checked by reading, not run on hardware
//
#include <hip/hip_runtime.h>

// ---------------- types ----------------
typedef __attribute__((ext_vector_type(16))) __bf16 v16bf;
typedef __attribute__((ext_vector_type(8)))  float  v8f;

#define EPSV 1e-5f

// B=64 T=512 V=17 C=2 C1=64 H=128  N=B*T=32768  NI=H*V=2176
// ws layout (bytes):
//   0        : stats f32[1280]
//   8192     : w2b  bf16[8192]    (swizzled B fragments for GCN2, K=64,N=128)
//   24576    : wtb  bf16[835584]  (swizzled B fragments for temporal conv, K=3*2176,N=128)
//   2097152  : gbuf bf16[64*512*2176]  (y2 / g in [b,t, hh*17+v] layout)

static __device__ __forceinline__ __bf16 f2bf(float f) {
  unsigned u = __builtin_bit_cast(unsigned, f);
  unsigned r = u + 0x7FFFu + ((u >> 16) & 1u);       // round-to-nearest-even
  unsigned short h = (unsigned short)(r >> 16);
  return __builtin_bit_cast(__bf16, h);
}
static __device__ __forceinline__ float bf2f(__bf16 b) {
  unsigned short s = __builtin_bit_cast(unsigned short, b);
  unsigned u = ((unsigned)s) << 16;
  return __builtin_bit_cast(float, u);
}

static __device__ __forceinline__ v8f wmma_bf16(v16bf a, v16bf b, v8f c) {
  // D = A(16x32 bf16) x B(32x16 bf16) + C(16x16 f32)
  return __builtin_amdgcn_wmma_f32_16x16x32_bf16(false, a, false, b, (short)0, c,
                                                 false, false);
}

// A-fragment gather per ISA 7.12.2 (16-bit A 16x32):
// lane<16: K = kbase+0..7 and kbase+16..23 ; lane>=16: +8 on both halves.
static __device__ __forceinline__ v16bf load_afrag(const __bf16* rowptr, int lane, int kbase) {
  int hi = lane >> 4;
  const __bf16* p0 = rowptr + kbase + hi * 8;
  union { v16bf v; uint4 q[2]; } u;
  u.q[0] = *(const uint4*)p0;
  u.q[1] = *(const uint4*)(p0 + 16);
  return u.v;
}

// CDNA5 async copy: 16B global -> LDS, tracked by ASYNCcnt (ISA 10/15.18).
static __device__ __forceinline__ void async_g2l_b128(unsigned lds_off, const void* gptr) {
  unsigned long long ga = (unsigned long long)(size_t)gptr;
  asm volatile("global_load_async_to_lds_b128 %0, %1, off"
               :: "v"(lds_off), "v"(ga) : "memory");
}
static __device__ __forceinline__ void wait_asynccnt0() {
  asm volatile("s_wait_asynccnt 0x0" ::: "memory");
}
// low 32 bits of a generic pointer into LDS == LDS byte offset (aperture map)
static __device__ __forceinline__ unsigned lds_off32(const void* p) {
  return (unsigned)(size_t)p;
}

// ---------------- K0: weight swizzle + stats zero ----------------
__global__ __launch_bounds__(256) void k_prep(const float* __restrict__ wt,
                                              const float* __restrict__ w2,
                                              __bf16* __restrict__ wtb,
                                              __bf16* __restrict__ w2b,
                                              float* __restrict__ stats) {
  int t = blockIdx.x * 256 + threadIdx.x;
  const int NW = 3 * 68 * 8 * 32 * 16;  // 835584
  if (t < NW) {
    // consumer index: (((k*68+ic)*8+nt)*32+lane)*16 + j
    int j = t & 15, lane = (t >> 4) & 31, nt = (t >> 9) & 7;
    int rest = t >> 12, ic = rest % 68, k = rest / 68;
    int col = nt * 16 + (lane & 15), hi = lane >> 4;
    int kk = ((j < 8) ? j : j + 8) + hi * 8;      // B-fragment K striping
    int i = ic * 32 + kk;
    wtb[t] = f2bf(wt[((size_t)col * 2176 + i) * 3 + k]);
  } else if (t < NW + 8192) {
    int u = t - NW;
    int j = u & 15, lane = (u >> 4) & 31, ks = (u >> 9) & 1, nt = u >> 10;
    int col = nt * 16 + (lane & 15), hi = lane >> 4;
    int kk = ((j < 8) ? j : j + 8) + hi * 8;
    int c = ks * 32 + kk;
    w2b[u] = f2bf(w2[col * 64 + c]);
  } else if (t < NW + 8192 + 1280) {
    stats[t - NW - 8192] = 0.f;
  }
}

// ---------------- K1: GCN1 pre-BN stats (tiny compute, VALU) ----------------
__global__ __launch_bounds__(256) void k_gcn1_stats(const float* __restrict__ x,
                                                    const float* __restrict__ adj,
                                                    const float* __restrict__ w1,
                                                    const float* __restrict__ b1,
                                                    float* __restrict__ stats) {
  __shared__ float adjs[289];
  __shared__ float bsum[64], bsq[64];
  int tid = threadIdx.x;
  if (tid < 289) adjs[tid] = adj[tid];
  if (tid < 64) { bsum[tid] = 0.f; bsq[tid] = 0.f; }
  __syncthreads();
  int n = blockIdx.x * 256 + tid;
  float xa[17][2];
  {
    float xv[17][2];
    const float* xp = x + (size_t)n * 34;
#pragma unroll
    for (int v = 0; v < 17; ++v) { xv[v][0] = xp[v * 2]; xv[v][1] = xp[v * 2 + 1]; }
#pragma unroll
    for (int vp = 0; vp < 17; ++vp) {
      float a0 = 0.f, a1 = 0.f;
#pragma unroll
      for (int v = 0; v < 17; ++v) {
        float a = adjs[v * 17 + vp];
        a0 += a * xv[v][0]; a1 += a * xv[v][1];
      }
      xa[vp][0] = a0; xa[vp][1] = a1;
    }
  }
  for (int o = 0; o < 64; ++o) {
    float wa = w1[o * 2], wb = w1[o * 2 + 1], bb = b1[o];
    float s = 0.f, q = 0.f;
#pragma unroll
    for (int v = 0; v < 17; ++v) {
      float y = xa[v][0] * wa + xa[v][1] * wb + bb;
      s += y; q += y * y;
    }
    atomicAdd(&bsum[o], s); atomicAdd(&bsq[o], q);
  }
  __syncthreads();
  if (tid < 64) { atomicAdd(&stats[tid], bsum[tid]); atomicAdd(&stats[64 + tid], bsq[tid]); }
}

// ---------------- finalize BN stats -> scale/shift ----------------
__global__ void k_finalize(const float* __restrict__ sum, const float* __restrict__ sq,
                           const float* __restrict__ gamma, const float* __restrict__ beta,
                           float* __restrict__ scale, float* __restrict__ shift,
                           int nch, float invCount) {
  int o = threadIdx.x;
  if (o < nch) {
    float m = sum[o] * invCount;
    float v = sq[o] * invCount - m * m;
    float sc = gamma[o] * rsqrtf(v + EPSV);
    scale[o] = sc;
    shift[o] = beta[o] - m * sc;
  }
}

// ---------------- K3: GCN2 via bf16 WMMA; writes y2 (pre-BN) to gbuf -------
__global__ __launch_bounds__(256) void k_gcn2_wmma(const float* __restrict__ x,
                                                   const float* __restrict__ adj,
                                                   const float* __restrict__ w1,
                                                   const float* __restrict__ b1,
                                                   const float* __restrict__ b2,
                                                   float* __restrict__ stats,
                                                   const __bf16* __restrict__ w2b,
                                                   __bf16* __restrict__ gbuf) {
  __shared__ float adjs[289];
  __shared__ float xls[8][34];
  __shared__ float xas[8][17][2];
  __shared__ __bf16 a2s[8][32][72];   // 72-stride: 144B rows -> bank stagger, 16B aligned
  __shared__ __bf16 w2bs[8192];
  __shared__ float colsum[128], colsq[128];
  int tid = threadIdx.x;
  // phase 0: stage inputs, zero pads
  if (tid < 289) adjs[tid] = adj[tid];
  {
    int n0 = blockIdx.x * 8;
    if (tid < 272) { int nl = tid / 34, j = tid % 34; xls[nl][j] = x[(size_t)(n0 + nl) * 34 + j]; }
  }
  // async copy of swizzled GCN2 weight fragments into LDS (ASYNCcnt path)
  for (int i4 = tid; i4 < 1024; i4 += 256)
    async_g2l_b128(lds_off32(w2bs) + i4 * 16, (const void*)((const uint4*)w2b + i4));
  for (int idx = tid; idx < 960; idx += 256) {   // zero A rows 17..31
    int w = idx / 120, rem = idx % 120, r = rem / 8, o4 = rem % 8;
    uint4 z; z.x = z.y = z.z = z.w = 0u;
    ((uint4*)&a2s[w][17 + r][0])[o4] = z;
  }
  if (tid < 128) { colsum[tid] = 0.f; colsq[tid] = 0.f; }
  wait_asynccnt0();
  __syncthreads();
  // phase 1a: adjacency-mix raw input (xa = adj^T x)
  if (tid < 136) {
    int nl = tid / 17, v = tid % 17;
    float a0 = 0.f, a1 = 0.f;
    for (int w = 0; w < 17; ++w) {
      float a = adjs[w * 17 + v];
      a0 += a * xls[nl][w * 2]; a1 += a * xls[nl][w * 2 + 1];
    }
    xas[nl][v][0] = a0; xas[nl][v][1] = a1;
  }
  __syncthreads();
  // phase 1b: h1 = relu(bn1(xa*w1+b1)); a2 = adj^T h1 -> bf16 A staging
  const float* scale1 = stats + 640;
  const float* shift1 = stats + 704;
  for (int task = tid; task < 512; task += 256) {
    int nl = task >> 6, o = task & 63;
    float wa = w1[o * 2], wb = w1[o * 2 + 1], bb = b1[o];
    float sc = scale1[o], sh = shift1[o];
    float h1[17];
#pragma unroll
    for (int v = 0; v < 17; ++v) {
      float y = xas[nl][v][0] * wa + xas[nl][v][1] * wb + bb;
      y = y * sc + sh;
      h1[v] = y > 0.f ? y : 0.f;
    }
#pragma unroll
    for (int v = 0; v < 17; ++v) {
      float a2 = 0.f;
#pragma unroll
      for (int w = 0; w < 17; ++w) a2 += adjs[w * 17 + v] * h1[w];
      a2s[nl][v][o] = f2bf(a2);
    }
  }
  __syncthreads();
  // phase 2: per-wave GEMM  y2[32(pad17) x 128] = a2[32x64] * w2^T[64x128]
  int wave = tid >> 5, lane = tid & 31;
  size_t n = (size_t)blockIdx.x * 8 + wave;
  for (int mt = 0; mt < 2; ++mt) {
    const __bf16* arow = &a2s[wave][mt * 16 + (lane & 15)][0];
    v16bf a0 = load_afrag(arow, lane, 0);
    v16bf a1 = load_afrag(arow, lane, 32);
    for (int nt = 0; nt < 8; ++nt) {
      v8f acc = {0.f, 0.f, 0.f, 0.f, 0.f, 0.f, 0.f, 0.f};
      v16bf bf0 = *(const v16bf*)&w2bs[((nt * 2 + 0) * 32 + lane) * 16];
      v16bf bf1 = *(const v16bf*)&w2bs[((nt * 2 + 1) * 32 + lane) * 16];
      acc = wmma_bf16(a0, bf0, acc);
      acc = wmma_bf16(a1, bf1, acc);
      int col = nt * 16 + (lane & 15);
      float bb = b2[col];
      int rbase = mt * 16 + ((lane >> 4) ? 8 : 0);
      float s = 0.f, q = 0.f;
#pragma unroll
      for (int r = 0; r < 8; ++r) {            // C/D layout: vgpr r -> row rbase+r
        int row = rbase + r;
        if (row < 17) {
          float z = acc[r] + bb;
          s += z; q += z * z;
          gbuf[n * 2176 + (size_t)col * 17 + row] = f2bf(z);
        }
      }
      if (rbase < 17) { atomicAdd(&colsum[col], s); atomicAdd(&colsq[col], q); }
    }
  }
  __syncthreads();
  if (tid < 128) { atomicAdd(&stats[128 + tid], colsum[tid]); atomicAdd(&stats[256 + tid], colsq[tid]); }
}

// ---------------- K4b: BN2 + ReLU in place on gbuf ----------------
__global__ __launch_bounds__(256) void k_bn2_relu(__bf16* __restrict__ gbuf,
                                                  const float* __restrict__ stats) {
  const float* sc = stats + 768;
  const float* sh = stats + 896;
  size_t t8 = (size_t)blockIdx.x * 256 + threadIdx.x;
  size_t base = t8 * 8;                    // 2176 % 8 == 0 -> never crosses a row
  uint4 pk = *(uint4*)((char*)gbuf + base * 2);
  __bf16* e = (__bf16*)&pk;
  int im = (int)(base % 2176);
#pragma unroll
  for (int j = 0; j < 8; ++j) {
    int hh = (im + j) / 17;                // BN2 channel
    float f = bf2f(e[j]) * sc[hh] + sh[hh];
    e[j] = f2bf(f > 0.f ? f : 0.f);
  }
  *(uint4*)((char*)gbuf + base * 2) = pk;
}

// ---------------- K5: temporal conv (k=3,pad=1) as bf16 WMMA GEMM ----------
// Block covers 32 t-rows x 128 o-cols (2 M-tiles per wave) to halve L2 weight
// traffic; A tile (34 rows x 2176 bf16, zero-padded T edges) staged via
// global_load_async_to_lds_b128.
__global__ __launch_bounds__(256) void k_tconv_wmma(const __bf16* __restrict__ gbuf,
                                                    const __bf16* __restrict__ wtb,
                                                    const float* __restrict__ bt,
                                                    float* __restrict__ out,
                                                    float* __restrict__ stats) {
  extern __shared__ char smem[];
  __bf16* smA = (__bf16*)smem;                      // 34 rows * 2184 bf16 (stagger pad)
  float* colsum = (float*)(smem + 148512);
  float* colsq = colsum + 128;
  int tid = threadIdx.x;
  int b = blockIdx.x >> 4;
  int t0 = (blockIdx.x & 15) << 5;
  if (tid < 128) { colsum[tid] = 0.f; colsq[tid] = 0.f; }
  // stage rows t0-1 .. t0+32 of g (zero-pad T boundary)
  unsigned smA0 = lds_off32(smA);
  for (int idx = tid; idx < 34 * 272; idx += 256) {
    int row = idx / 272, off = idx % 272;
    int t = t0 - 1 + row;
    unsigned dst = smA0 + (unsigned)row * 4368u + (unsigned)off * 16u;
    if (t >= 0 && t < 512) {
      async_g2l_b128(dst, (const void*)(((const uint4*)(gbuf + ((size_t)b * 512 + t) * 2176)) + off));
    } else {
      uint4 z; z.x = z.y = z.z = z.w = 0u;
      ((uint4*)(smA + (size_t)row * 2184))[off] = z;
    }
  }
  wait_asynccnt0();
  __syncthreads();
  int wave = tid >> 5, lane = tid & 31;
  int nt = wave;                                    // one 16-col o-tile per wave
  v8f acc0 = {0.f, 0.f, 0.f, 0.f, 0.f, 0.f, 0.f, 0.f};
  v8f acc1 = {0.f, 0.f, 0.f, 0.f, 0.f, 0.f, 0.f, 0.f};
  const __bf16* rowbase = smA + (size_t)(lane & 15) * 2184;
  for (int k = 0; k < 3; ++k) {
    const __bf16* rp0 = rowbase + (size_t)k * 2184;      // rows t0..t0+15 (+k)
    const __bf16* rp1 = rp0 + (size_t)16 * 2184;          // rows t0+16..t0+31 (+k)
    const __bf16* wb = wtb + ((size_t)(k * 68) * 8 + nt) * 512 + (size_t)lane * 16;
    for (int ic = 0; ic < 68; ++ic) {
      __builtin_prefetch(wb + 4096, 0, 0);          // next B fragment (L2 stream)
      v16bf bfrag = *(const v16bf*)wb;              // B reused by both M-tiles
      v16bf a0 = load_afrag(rp0, lane, ic * 32);
      v16bf a1 = load_afrag(rp1, lane, ic * 32);
      acc0 = wmma_bf16(a0, bfrag, acc0);
      acc1 = wmma_bf16(a1, bfrag, acc1);
      wb += 4096;                                   // 8 ntiles * 512
    }
  }
  int col = nt * 16 + (lane & 15);
  float bb = bt[col];
  int rbase = (lane >> 4) ? 8 : 0;
  float s = 0.f, q = 0.f;
#pragma unroll
  for (int r = 0; r < 8; ++r) {
    int t = t0 + rbase + r;
    float z = acc0[r] + bb;
    s += z; q += z * z;
    out[((size_t)b * 512 + t) * 128 + col] = z;     // pre-BN z into d_out
  }
#pragma unroll
  for (int r = 0; r < 8; ++r) {
    int t = t0 + 16 + rbase + r;
    float z = acc1[r] + bb;
    s += z; q += z * z;
    out[((size_t)b * 512 + t) * 128 + col] = z;
  }
  atomicAdd(&colsum[col], s);
  atomicAdd(&colsq[col], q);
  __syncthreads();
  if (tid < 128) { atomicAdd(&stats[384 + tid], colsum[tid]); atomicAdd(&stats[512 + tid], colsq[tid]); }
}

// ---------------- K7: BN3 + ReLU in place on d_out ----------------
__global__ __launch_bounds__(256) void k_bn3_relu(float* __restrict__ out,
                                                  const float* __restrict__ stats) {
  const float* sc = stats + 1024;
  const float* sh = stats + 1152;
  size_t idx = (size_t)blockIdx.x * 256 + threadIdx.x;
  int col = (int)(idx & 127);
  float z = out[idx] * sc[col] + sh[col];
  out[idx] = z > 0.f ? z : 0.f;
}

// ---------------- launch ----------------
extern "C" void kernel_launch(void* const* d_in, const int* in_sizes, int n_in,
                              void* d_out, int out_size, void* d_ws, size_t ws_size,
                              hipStream_t stream) {
  const float* x   = (const float*)d_in[0];
  const float* adj = (const float*)d_in[1];
  const float* w1  = (const float*)d_in[2];
  const float* b1  = (const float*)d_in[3];
  const float* g1  = (const float*)d_in[4];
  const float* be1 = (const float*)d_in[5];
  const float* w2  = (const float*)d_in[6];
  const float* b2  = (const float*)d_in[7];
  const float* g2  = (const float*)d_in[8];
  const float* be2 = (const float*)d_in[9];
  const float* wt  = (const float*)d_in[10];
  const float* bt  = (const float*)d_in[11];
  const float* gt  = (const float*)d_in[12];
  const float* bet = (const float*)d_in[13];
  (void)in_sizes; (void)n_in; (void)out_size; (void)ws_size;

  char* ws = (char*)d_ws;
  float*  stats = (float*)ws;
  __bf16* w2b   = (__bf16*)(ws + 8192);
  __bf16* wtb   = (__bf16*)(ws + 24576);
  __bf16* gbuf  = (__bf16*)(ws + (1u << 21));
  float*  out   = (float*)d_out;

  k_prep<<<3301, 256, 0, stream>>>(wt, w2, wtb, w2b, stats);
  k_gcn1_stats<<<128, 256, 0, stream>>>(x, adj, w1, b1, stats);
  k_finalize<<<1, 64, 0, stream>>>(stats, stats + 64, g1, be1,
                                   stats + 640, stats + 704, 64, 1.f / 557056.f);
  k_gcn2_wmma<<<4096, 256, 0, stream>>>(x, adj, w1, b1, b2, stats, w2b, gbuf);
  k_finalize<<<1, 128, 0, stream>>>(stats + 128, stats + 256, g2, be2,
                                    stats + 768, stats + 896, 128, 1.f / 557056.f);
  k_bn2_relu<<<34816, 256, 0, stream>>>(gbuf, stats);
  k_tconv_wmma<<<1024, 256, 149536, stream>>>(gbuf, wtb, bt, out, stats);
  k_finalize<<<1, 128, 0, stream>>>(stats + 384, stats + 512, gt, bet,
                                    stats + 1024, stats + 1152, 128, 1.f / 32768.f);
  k_bn3_relu<<<16384, 256, 0, stream>>>(out, stats);
}
